// NeuralTuringMachine_84490596647313
// MI455X (gfx1250) — matrix operand order repaired and checked by reading, not compiled, not run on hardware
//
#include <hip/hip_runtime.h>
#include <hip/hip_bf16.h>
#include <math.h>

// ---------------------------------------------------------------------------
// NTM forward, MI455X (gfx1250, wave32, WMMA).
// memory==0  =>  read_vec==0  =>  out = h @ W_out[:, :H].T + b_out
// h = sigmoid(o) * tanh(sigmoid(i) * tanh(g)); f-gate unused.
// Pipeline: f32->bf16 convert/pack, GEMM1 (x @ W_ih[i|g|o].T fused LSTM act),
// GEMM2 (h @ W_out[:, :H].T + b_out). f32 accumulation via
// v_wmma_f32_16x16x32_bf16. Global->LDS staging uses the CDNA5 async-to-LDS
// path (ASYNCcnt) when the builtin exists; register staging otherwise.
// ---------------------------------------------------------------------------

typedef __bf16 v16bf __attribute__((ext_vector_type(16)));
typedef __bf16 v8bf  __attribute__((ext_vector_type(8)));
typedef __bf16 v4bf  __attribute__((ext_vector_type(4)));
typedef float  v8f   __attribute__((ext_vector_type(8)));
typedef int    v4i   __attribute__((ext_vector_type(4)));

typedef __attribute__((address_space(1))) void  as1_void;
typedef __attribute__((address_space(3))) void  as3_void;
typedef __attribute__((address_space(1))) v4i   as1_v4i;
typedef __attribute__((address_space(3))) v4i   as3_v4i;

constexpr int BATCH = 8192;
constexpr int DIN   = 1024;
constexpr int HH    = 2048;

constexpr int KT  = 32;        // WMMA K per step
constexpr int SA  = KT + 8;    // LDS row stride in bf16 (padding: 20-bank step)
constexpr int A_ELEMS = 128 * SA;   // A tile: 128 rows

#if __has_builtin(__builtin_amdgcn_global_load_async_to_lds_b128)
#define HAVE_ASYNC_LDS 1
#else
#define HAVE_ASYNC_LDS 0
#endif

__device__ __forceinline__ void async_cp16(const __bf16* g, __bf16* l) {
#if HAVE_ASYNC_LDS
  // two-step casts: addrspacecast (generic -> AS1/AS3), then pointee cast.
  as1_void* g1 = (as1_void*)(__bf16*)g;
  as3_void* l3 = (as3_void*)l;
  __builtin_amdgcn_global_load_async_to_lds_b128(
      (as1_v4i*)g1, (as3_v4i*)l3, 0, 0);
#else
  (void)g; (void)l;
#endif
}

__device__ __forceinline__ void wait_async_all() {
#if HAVE_ASYNC_LDS
#if __has_builtin(__builtin_amdgcn_s_wait_asynccnt)
  __builtin_amdgcn_s_wait_asynccnt(0);
#else
  asm volatile("s_wait_asynccnt 0x0" ::: "memory");
#endif
#endif
}

__device__ __forceinline__ float sigm(float x) {
  return 1.0f / (1.0f + __expf(-x));
}

// A fragment (16x32 bf16, MxK): lane<16 -> M=lane, K = 0..7 and 16..23;
// lane>=16 -> M=lane-16, K = 8..15 and 24..31.  (ISA 7.12.2 table)
__device__ __forceinline__ v16bf ld_frag_a(const __bf16* row0, int lane) {
  const __bf16* p = row0 + ((lane & 16) ? 8 : 0);
  v8bf lo = *(const v8bf*)(p);        // K = kb+0..7
  v8bf hi = *(const v8bf*)(p + 16);   // K = kb+16..23
  v16bf f;
#pragma unroll
  for (int e = 0; e < 8; ++e) { f[e] = lo[e]; f[e + 8] = hi[e]; }
  return f;
}

// B fragment (32x16 bf16, KxN): lane = N; lane<16 holds K=0..7 and 16..23,
// lane>=16 holds K=8..15 and 24..31 (same K order as A).
__device__ __forceinline__ v16bf ld_frag_b(const __bf16* col0, int lane) {
  const __bf16* p = col0 + ((lane & 16) ? 8 : 0);
  v8bf lo = *(const v8bf*)(p);
  v8bf hi = *(const v8bf*)(p + 16);
  v16bf f;
#pragma unroll
  for (int e = 0; e < 8; ++e) { f[e] = lo[e]; f[e + 8] = hi[e]; }
  return f;
}

// ---------------------------------------------------------------------------
// f32 -> bf16 (n divisible by 4)
__global__ __launch_bounds__(256) void cvt_f32_bf16(
    const float* __restrict__ s, __bf16* __restrict__ d, long n) {
  long i = ((long)blockIdx.x * blockDim.x + threadIdx.x) * 4;
  if (i >= n) return;
  float4 v = *(const float4*)(s + i);
  v4bf o;
  o[0] = (__bf16)v.x; o[1] = (__bf16)v.y; o[2] = (__bf16)v.z; o[3] = (__bf16)v.w;
  *(v4bf*)(d + i) = o;
}

// Pack W_out (1024 x 2112) cols [0,2048) -> bf16 (1024 x 2048)
__global__ __launch_bounds__(256) void pack_wout(
    const float* __restrict__ w, __bf16* __restrict__ d) {
  long i = ((long)blockIdx.x * blockDim.x + threadIdx.x) * 4; // over 1024*2048
  int row = (int)(i >> 11);
  int col = (int)(i & 2047);
  float4 v = *(const float4*)(w + (long)row * (HH + 64) + col);
  v4bf o;
  o[0] = (__bf16)v.x; o[1] = (__bf16)v.y; o[2] = (__bf16)v.z; o[3] = (__bf16)v.w;
  *(v4bf*)(d + i) = o;
}

// ---------------------------------------------------------------------------
// GEMM1: h[b, n] = act over (x @ W_ih[{i,g,o}].T + bias), block 128x64,
// 8 waves (4x2), wave tile 32x32, K-loop over DIN in steps of 32.
constexpr int B1_ELEMS = 3 * 64 * SA;
constexpr int BUF1     = A_ELEMS + B1_ELEMS;

__global__ __launch_bounds__(256, 1) void gemm_gates(
    const __bf16* __restrict__ xb, const __bf16* __restrict__ wb,
    const float* __restrict__ b_ih, const float* __restrict__ b_hh,
    __bf16* __restrict__ hb) {
  __shared__ __align__(16) __bf16 sm[2][BUF1];

  const int bm = blockIdx.x, bn = blockIdx.y;
  const int t = threadIdx.x;
  const int wave = t >> 5, lane = t & 31;
  const int wm = wave >> 1, wn = wave & 1;
  const int fr = lane & 15;

  const int secoff[3] = {0, 2 * HH, 3 * HH};

  const v8f zf = {0.f, 0.f, 0.f, 0.f, 0.f, 0.f, 0.f, 0.f};
  v8f acc[3][2][2];
#pragma unroll
  for (int g = 0; g < 3; ++g)
#pragma unroll
    for (int i = 0; i < 2; ++i)
#pragma unroll
      for (int j = 0; j < 2; ++j) acc[g][i][j] = zf;

  // global staging: 16B chunks. A: 512 chunks (2/thread), B: 3*256 (3/thread)
  const int arow0 = t >> 2;
  const int arow1 = (t + 256) >> 2;
  const int aoff  = (t & 3) * 8;
  const __bf16* gA0 = xb + (size_t)(bm * 128 + arow0) * DIN + aoff;
  const __bf16* gA1 = xb + (size_t)(bm * 128 + arow1) * DIN + aoff;
  const __bf16* gB[3];
#pragma unroll
  for (int g = 0; g < 3; ++g)
    gB[g] = wb + (size_t)(secoff[g] + bn * 64 + arow0) * DIN + aoff;

  constexpr int NK = DIN / KT;  // 32

#if HAVE_ASYNC_LDS
  auto stage = [&](int k, int buf) {
    __bf16* s = sm[buf];
    const int ko = k * KT;
    async_cp16(gA0 + ko, s + arow0 * SA + aoff);
    async_cp16(gA1 + ko, s + arow1 * SA + aoff);
#pragma unroll
    for (int g = 0; g < 3; ++g)
      async_cp16(gB[g] + ko, s + A_ELEMS + (g * 64 + arow0) * SA + aoff);
  };
  stage(0, 0);
  wait_async_all();
  __syncthreads();
#else
  uint4 ra0, ra1, rb[3];
  auto load_regs = [&](int k) {
    const int ko = k * KT;
    ra0 = *(const uint4*)(gA0 + ko);
    ra1 = *(const uint4*)(gA1 + ko);
#pragma unroll
    for (int g = 0; g < 3; ++g) rb[g] = *(const uint4*)(gB[g] + ko);
  };
  auto store_lds = [&](int buf) {
    __bf16* s = sm[buf];
    *(uint4*)(s + arow0 * SA + aoff) = ra0;
    *(uint4*)(s + arow1 * SA + aoff) = ra1;
#pragma unroll
    for (int g = 0; g < 3; ++g)
      *(uint4*)(s + A_ELEMS + (g * 64 + arow0) * SA + aoff) = rb[g];
  };
  load_regs(0);
  store_lds(0);
  __syncthreads();
#endif

  for (int k = 0; k < NK; ++k) {
#if HAVE_ASYNC_LDS
    if (k + 1 < NK) stage(k + 1, (k + 1) & 1);
#else
    if (k + 1 < NK) load_regs(k + 1);
#endif
    if (k + 2 < NK) {
      __builtin_prefetch(gA0 + (k + 2) * KT, 0, 3);
      __builtin_prefetch(gB[0] + (k + 2) * KT, 0, 3);
    }
    const __bf16* s = sm[k & 1];
    v16bf fa[2], fb[3][2];
#pragma unroll
    for (int i = 0; i < 2; ++i)
      fa[i] = ld_frag_a(s + (wm * 32 + i * 16 + fr) * SA, lane);
#pragma unroll
    for (int g = 0; g < 3; ++g)
#pragma unroll
      for (int j = 0; j < 2; ++j)
        fb[g][j] = ld_frag_b(s + A_ELEMS + (g * 64 + wn * 32 + j * 16 + fr) * SA, lane);
#pragma unroll
    for (int g = 0; g < 3; ++g)
#pragma unroll
      for (int i = 0; i < 2; ++i)
#pragma unroll
        for (int j = 0; j < 2; ++j)
          acc[g][i][j] = __builtin_amdgcn_wmma_f32_16x16x32_bf16(
              false, fa[i], false, fb[g][j], (short)0, acc[g][i][j], false, false);
#if HAVE_ASYNC_LDS
    if (k + 1 < NK) wait_async_all();
#else
    if (k + 1 < NK) store_lds((k + 1) & 1);
#endif
    __syncthreads();
  }

  // fused LSTM-cell epilogue -> bf16 h
  const int rh = (lane >> 4) * 8;
#pragma unroll
  for (int i = 0; i < 2; ++i)
#pragma unroll
    for (int j = 0; j < 2; ++j) {
      const int gcol = bn * 64 + wn * 32 + j * 16 + (lane & 15);
      const float bi = b_ih[gcol] + b_hh[gcol];
      const float bg = b_ih[2 * HH + gcol] + b_hh[2 * HH + gcol];
      const float bo = b_ih[3 * HH + gcol] + b_hh[3 * HH + gcol];
#pragma unroll
      for (int v = 0; v < 8; ++v) {
        const int grow = bm * 128 + wm * 32 + i * 16 + rh + v;
        const float iv = acc[0][i][j][v] + bi;
        const float gv = acc[1][i][j][v] + bg;
        const float ov = acc[2][i][j][v] + bo;
        const float cc = sigm(iv) * tanhf(gv);
        const float hv = sigm(ov) * tanhf(cc);
        hb[(size_t)grow * HH + gcol] = (__bf16)hv;
      }
    }
}

// ---------------------------------------------------------------------------
// GEMM2: out = h @ W_out[:, :H].T + b_out, block 128x64, K over HH.
constexpr int B2_ELEMS = 64 * SA;
constexpr int BUF2     = A_ELEMS + B2_ELEMS;

__global__ __launch_bounds__(256, 1) void gemm_out(
    const __bf16* __restrict__ hb, const __bf16* __restrict__ wob,
    const float* __restrict__ b_out, float* __restrict__ out) {
  __shared__ __align__(16) __bf16 sm[2][BUF2];

  const int bm = blockIdx.x, bn = blockIdx.y;
  const int t = threadIdx.x;
  const int wave = t >> 5, lane = t & 31;
  const int wm = wave >> 1, wn = wave & 1;
  const int fr = lane & 15;

  const v8f zf = {0.f, 0.f, 0.f, 0.f, 0.f, 0.f, 0.f, 0.f};
  v8f acc[2][2];
#pragma unroll
  for (int i = 0; i < 2; ++i)
#pragma unroll
    for (int j = 0; j < 2; ++j) acc[i][j] = zf;

  const int arow0 = t >> 2;
  const int arow1 = (t + 256) >> 2;
  const int aoff  = (t & 3) * 8;
  const __bf16* gA0 = hb + (size_t)(bm * 128 + arow0) * HH + aoff;
  const __bf16* gA1 = hb + (size_t)(bm * 128 + arow1) * HH + aoff;
  const __bf16* gB0 = wob + (size_t)(bn * 64 + arow0) * HH + aoff;

  constexpr int NK = HH / KT;  // 64

#if HAVE_ASYNC_LDS
  auto stage = [&](int k, int buf) {
    __bf16* s = sm[buf];
    const int ko = k * KT;
    async_cp16(gA0 + ko, s + arow0 * SA + aoff);
    async_cp16(gA1 + ko, s + arow1 * SA + aoff);
    async_cp16(gB0 + ko, s + A_ELEMS + arow0 * SA + aoff);
  };
  stage(0, 0);
  wait_async_all();
  __syncthreads();
#else
  uint4 ra0, ra1, rb0;
  auto load_regs = [&](int k) {
    const int ko = k * KT;
    ra0 = *(const uint4*)(gA0 + ko);
    ra1 = *(const uint4*)(gA1 + ko);
    rb0 = *(const uint4*)(gB0 + ko);
  };
  auto store_lds = [&](int buf) {
    __bf16* s = sm[buf];
    *(uint4*)(s + arow0 * SA + aoff) = ra0;
    *(uint4*)(s + arow1 * SA + aoff) = ra1;
    *(uint4*)(s + A_ELEMS + arow0 * SA + aoff) = rb0;
  };
  load_regs(0);
  store_lds(0);
  __syncthreads();
#endif

  for (int k = 0; k < NK; ++k) {
#if HAVE_ASYNC_LDS
    if (k + 1 < NK) stage(k + 1, (k + 1) & 1);
#else
    if (k + 1 < NK) load_regs(k + 1);
#endif
    if (k + 2 < NK) {
      __builtin_prefetch(gA0 + (k + 2) * KT, 0, 3);
      __builtin_prefetch(gB0 + (k + 2) * KT, 0, 3);
    }
    const __bf16* s = sm[k & 1];
    v16bf fa[2], fb[2];
#pragma unroll
    for (int i = 0; i < 2; ++i)
      fa[i] = ld_frag_a(s + (wm * 32 + i * 16 + fr) * SA, lane);
#pragma unroll
    for (int j = 0; j < 2; ++j)
      fb[j] = ld_frag_b(s + A_ELEMS + (wn * 32 + j * 16 + fr) * SA, lane);
#pragma unroll
    for (int i = 0; i < 2; ++i)
#pragma unroll
      for (int j = 0; j < 2; ++j)
        acc[i][j] = __builtin_amdgcn_wmma_f32_16x16x32_bf16(
            false, fa[i], false, fb[j], (short)0, acc[i][j], false, false);
#if HAVE_ASYNC_LDS
    if (k + 1 < NK) wait_async_all();
#else
    if (k + 1 < NK) store_lds((k + 1) & 1);
#endif
    __syncthreads();
  }

  const int rh = (lane >> 4) * 8;
#pragma unroll
  for (int i = 0; i < 2; ++i)
#pragma unroll
    for (int j = 0; j < 2; ++j) {
      const int gcol = bn * 64 + wn * 32 + j * 16 + (lane & 15);
      const float bo = b_out[gcol];
#pragma unroll
      for (int v = 0; v < 8; ++v) {
        const int grow = bm * 128 + wm * 32 + i * 16 + rh + v;
        out[(size_t)grow * DIN + gcol] = acc[i][j][v] + bo;
      }
    }
}

// ---------------------------------------------------------------------------
extern "C" void kernel_launch(void* const* d_in, const int* in_sizes, int n_in,
                              void* d_out, int out_size, void* d_ws, size_t ws_size,
                              hipStream_t stream) {
  (void)in_sizes; (void)n_in; (void)out_size; (void)ws_size;
  const float* x     = (const float*)d_in[0];
  const float* W_ih  = (const float*)d_in[1];
  const float* b_ih  = (const float*)d_in[2];
  const float* b_hh  = (const float*)d_in[3];
  // d_in[4] = W_read, d_in[5] = b_read: provably dead (memory == 0 -> read_vec == 0)
  const float* W_out = (const float*)d_in[6];
  const float* b_out = (const float*)d_in[7];
  float* out = (float*)d_out;

  // workspace layout (bf16 elems): xb | W_ih(all 4H) | W_out packed | h
  __bf16* xb    = (__bf16*)d_ws;
  __bf16* wihb  = xb    + (size_t)BATCH * DIN;        // 8192*1024
  __bf16* woutb = wihb  + (size_t)4 * HH * DIN;       // 8192*1024
  __bf16* hb    = woutb + (size_t)DIN * HH;           // 1024*2048

  cvt_f32_bf16<<<(BATCH * DIN) / 4 / 256, 256, 0, stream>>>(x, xb, (long)BATCH * DIN);
  cvt_f32_bf16<<<(4 * HH * DIN) / 4 / 256, 256, 0, stream>>>(W_ih, wihb, (long)4 * HH * DIN);
  pack_wout<<<(DIN * HH) / 4 / 256, 256, 0, stream>>>(W_out, woutb);

  gemm_gates<<<dim3(BATCH / 128, HH / 64), 256, 0, stream>>>(xb, wihb, b_ih, b_hh, hb);
  gemm_out<<<dim3(BATCH / 128, DIN / 64), 256, 0, stream>>>(hb, woutb, b_out, out);
}
